// HWxHW_TransformerBlock_33990371181535
// MI455X (gfx1250) — matrix operand order repaired and checked by reading, not compiled, not run on hardware
//
#include <hip/hip_runtime.h>
#include <hip/hip_bf16.h>
#include <math.h>

// ---------- types for WMMA ----------
typedef __attribute__((ext_vector_type(16))) _Float16 v16h;
typedef __attribute__((ext_vector_type(8)))  _Float16 v8h;
typedef __attribute__((ext_vector_type(8)))  float    v8f;

// ---------- problem constants ----------
#define BATCH 2
#define DIM   48
#define HEADS 2
#define CHEAD 24
#define CPAD  32
#define HW    64
#define NPIX  4096           // 64*64
#define C3    144            // DIM*3
#define HID   96
#define HID2  192
#define BH    (BATCH*HEADS)  // 4

// =====================================================================
// LayerNorm over channel dim (NCHW), per (b, pixel)
// =====================================================================
__global__ void ln_kernel(const float* __restrict__ x,
                          const float* __restrict__ w,
                          const float* __restrict__ b,
                          float* __restrict__ out) {
    int t = blockIdx.x * blockDim.x + threadIdx.x;
    if (t >= BATCH * NPIX) return;
    int bb = t >> 12;
    int p  = t & (NPIX - 1);
    const float* xp = x + (size_t)bb * DIM * NPIX + p;
    float s = 0.f, s2 = 0.f;
    #pragma unroll 4
    for (int c = 0; c < DIM; ++c) {
        float v = xp[c * NPIX];
        s += v; s2 += v * v;
    }
    float mu  = s * (1.0f / DIM);
    float var = s2 * (1.0f / DIM) - mu * mu;
    float inv = rsqrtf(var + 1e-5f);
    float* op = out + (size_t)bb * DIM * NPIX + p;
    #pragma unroll 4
    for (int c = 0; c < DIM; ++c) {
        op[c * NPIX] = (xp[c * NPIX] - mu) * inv * w[c] + b[c];
    }
}

// =====================================================================
// 1x1 conv (pointwise GEMM over channels), optional residual add
// =====================================================================
__global__ void conv1x1_kernel(const float* __restrict__ in,
                               const float* __restrict__ w,
                               const float* __restrict__ bias,
                               const float* __restrict__ resid,
                               float* __restrict__ out,
                               int Cin, int Cout, int total) {
    int t = blockIdx.x * blockDim.x + threadIdx.x;
    if (t >= total) return;
    int p = t & (NPIX - 1);
    int o = (t >> 12) % Cout;
    int b = t / (Cout << 12);
    const float* ip = in + ((size_t)b * Cin) * NPIX + p;
    const float* wp = w + (size_t)o * Cin;
    float acc = bias[o];
    #pragma unroll 4
    for (int c = 0; c < Cin; ++c) acc = fmaf(ip[c * NPIX], wp[c], acc);
    if (resid) acc += resid[t];
    out[t] = acc;
}

// =====================================================================
// 3x3 depthwise conv, SAME (zero pad)
// =====================================================================
__global__ void dw3_kernel(const float* __restrict__ in,
                           const float* __restrict__ w,
                           const float* __restrict__ bias,
                           float* __restrict__ out,
                           int Cc, int total) {
    int t = blockIdx.x * blockDim.x + threadIdx.x;
    if (t >= total) return;
    int p  = t & (NPIX - 1);
    int y  = p >> 6;
    int x  = p & 63;
    int ch = (t >> 12) % Cc;
    int b  = t / (Cc << 12);
    const float* ip = in + ((size_t)b * Cc + ch) * NPIX;
    const float* wp = w + (size_t)ch * 9;
    float acc = bias[ch];
    #pragma unroll
    for (int dy = -1; dy <= 1; ++dy) {
        int yy = y + dy;
        if (yy < 0 || yy >= HW) continue;
        #pragma unroll
        for (int dx = -1; dx <= 1; ++dx) {
            int xx = x + dx;
            if (xx < 0 || xx >= HW) continue;
            acc = fmaf(ip[(yy << 6) + xx], wp[(dy + 1) * 3 + (dx + 1)], acc);
        }
    }
    out[t] = acc;
}

// =====================================================================
// l2norm(q,k over chead) + pack:
//   Qp,Kp : [bh][N][32] f16 row-major (chead padded 24->32 with zeros)
//   Vt    : [bh][32][N] f16 (chead-major, rows 24..31 zero)
// =====================================================================
__global__ void pack_kernel(const float* __restrict__ qkvd,
                            _Float16* __restrict__ qp,
                            _Float16* __restrict__ kp,
                            _Float16* __restrict__ vt) {
    int t = blockIdx.x * blockDim.x + threadIdx.x;
    if (t >= BH * NPIX) return;
    int bh = t >> 12;
    int p  = t & (NPIX - 1);
    int b  = bh >> 1;
    int hd = bh & 1;
    const float* base = qkvd + (size_t)b * C3 * NPIX + p;

    // q
    float qv[CHEAD]; float ss = 0.f;
    #pragma unroll
    for (int c = 0; c < CHEAD; ++c) {
        qv[c] = base[(hd * CHEAD + c) * NPIX];
        ss += qv[c] * qv[c];
    }
    float inv = 1.0f / fmaxf(sqrtf(ss), 1e-12f);
    _Float16* qrow = qp + ((size_t)bh * NPIX + p) * CPAD;
    #pragma unroll
    for (int c = 0; c < CHEAD; ++c) qrow[c] = (_Float16)(qv[c] * inv);
    #pragma unroll
    for (int c = CHEAD; c < CPAD; ++c) qrow[c] = (_Float16)0.f;

    // k
    ss = 0.f;
    #pragma unroll
    for (int c = 0; c < CHEAD; ++c) {
        qv[c] = base[(DIM + hd * CHEAD + c) * NPIX];
        ss += qv[c] * qv[c];
    }
    inv = 1.0f / fmaxf(sqrtf(ss), 1e-12f);
    _Float16* krow = kp + ((size_t)bh * NPIX + p) * CPAD;
    #pragma unroll
    for (int c = 0; c < CHEAD; ++c) krow[c] = (_Float16)(qv[c] * inv);
    #pragma unroll
    for (int c = CHEAD; c < CPAD; ++c) krow[c] = (_Float16)0.f;

    // v transposed
    #pragma unroll
    for (int c = 0; c < CHEAD; ++c)
        vt[((size_t)bh * CPAD + c) * NPIX + p] = (_Float16)base[(2 * DIM + hd * CHEAD + c) * NPIX];
    #pragma unroll
    for (int c = CHEAD; c < CPAD; ++c)
        vt[((size_t)bh * CPAD + c) * NPIX + p] = (_Float16)0.f;
}

// =====================================================================
// Flash attention, WMMA f32_16x16x32_f16.
// Grid: (32 q-tiles of 128, BH). Block: 256 threads = 8 waves.
// Wave w owns query rows [qtile*128 + w*16, +16).
// Softmax work is split across the half-wave pair (lane, lane^16):
// each lane handles 16 of the 32 key columns of its row; partial
// max/sum merged with one __shfl_xor so both lanes keep identical m,l.
// =====================================================================
__global__ __launch_bounds__(256) void attn_kernel(const _Float16* __restrict__ qp,
                                                   const _Float16* __restrict__ kp,
                                                   const _Float16* __restrict__ vt,
                                                   const float* __restrict__ temperature,
                                                   float* __restrict__ attn_out) {
    __shared__ float    sS[8][16][33];     // per-wave S scratch (padded)
    __shared__ _Float16 sP[8][16][32];     // per-wave P (f16)
    __shared__ float    sAlpha[8][16];
    __shared__ float    sL[8][16];

    const int tid  = threadIdx.x;
    const int wv   = tid >> 5;
    const int lane = tid & 31;
    const int hi   = lane >> 4;     // half-wave select
    const int ln16 = lane & 15;
    const int bh   = blockIdx.y;
    const int hd   = bh & 1;
    const float temp = temperature[hd];

    const int qrow  = blockIdx.x * 128 + wv * 16 + ln16;
    const int cbase = hi * 8;       // A-fragment column base per ISA layout
    const int jbase = hi * 16;      // softmax column half owned by this lane

    // ---- Q A-fragment: lanes 0-15 cols {0..15}, lanes 16-31 {16..31}
    const _Float16* qrowp = qp + ((size_t)bh * NPIX + qrow) * CPAD;
    v8h qlo = *(const v8h*)(qrowp + jbase);
    v8h qhi = *(const v8h*)(qrowp + jbase + 8);
    v16h afrag = __builtin_shufflevector(qlo, qhi, 0,1,2,3,4,5,6,7,8,9,10,11,12,13,14,15);

    v8f O0 = {}; v8f O1 = {};
    float m = -3.0e38f;
    float l = 0.f;
    const int myrow = ln16;   // stats row (pair lane^16 keeps identical stats)

    for (int kc = 0; kc < NPIX / 32; ++kc) {
        const int kb = kc * 32;

        // ---- S = Q * K^T for 16q x 32k : two WMMAs
        const _Float16* k0p = kp + ((size_t)bh * NPIX + kb + ln16) * CPAD + hi * 16;
        const _Float16* k1p = kp + ((size_t)bh * NPIX + kb + 16 + ln16) * CPAD + hi * 16;
        v16h bk0 = *(const v16h*)k0p;
        v16h bk1 = *(const v16h*)k1p;
        v8f z = {};
        v8f s0 = __builtin_amdgcn_wmma_f32_16x16x32_f16(false, afrag, false, bk0, (short)0, z, false, false);
        v8f s1 = __builtin_amdgcn_wmma_f32_16x16x32_f16(false, afrag, false, bk1, (short)0, z, false, false);

        // ---- spill S to per-wave LDS (C layout: row = r + 8*hi, col = ln16)
        #pragma unroll
        for (int r = 0; r < 8; ++r) {
            sS[wv][r + hi * 8][ln16]      = s0[r];
            sS[wv][r + hi * 8][16 + ln16] = s1[r];
        }
        asm volatile("s_wait_dscnt 0x0" ::: "memory");

        // ---- online softmax on row `myrow`: 16 columns per lane, merged via shfl
        float sv[16];
        float rmax = -3.0e38f;
        #pragma unroll
        for (int j = 0; j < 16; ++j) {
            sv[j] = sS[wv][myrow][jbase + j] * temp;
            rmax = fmaxf(rmax, sv[j]);
        }
        rmax = fmaxf(rmax, __shfl_xor(rmax, 16, 32));
        float mnew  = fmaxf(m, rmax);
        float alpha = __expf(m - mnew);
        float rsum  = 0.f;
        #pragma unroll
        for (int j = 0; j < 16; ++j) {
            float pv = __expf(sv[j] - mnew);
            rsum += pv;
            sP[wv][myrow][jbase + j] = (_Float16)pv;
        }
        rsum += __shfl_xor(rsum, 16, 32);
        l = l * alpha + rsum;
        m = mnew;
        sAlpha[wv][myrow] = alpha;   // both halves write identical value
        asm volatile("s_wait_dscnt 0x0" ::: "memory");

        // ---- rescale accumulators
        #pragma unroll
        for (int r = 0; r < 8; ++r) {
            float a = sAlpha[wv][r + hi * 8];
            O0[r] *= a;
            O1[r] *= a;
        }

        // ---- P A-fragment from LDS
        const _Float16* prow = &sP[wv][ln16][0];
        v8h plo = *(const v8h*)(prow + jbase);
        v8h phi = *(const v8h*)(prow + jbase + 8);
        v16h pfrag = __builtin_shufflevector(plo, phi, 0,1,2,3,4,5,6,7,8,9,10,11,12,13,14,15);

        // ---- V B-fragments (Vt rows are channels; contiguous in key dim)
        const _Float16* v0p = vt + ((size_t)bh * CPAD + ln16) * NPIX + kb + hi * 16;
        const _Float16* v1p = vt + ((size_t)bh * CPAD + 16 + ln16) * NPIX + kb + hi * 16;
        v16h bv0 = *(const v16h*)v0p;
        v16h bv1 = *(const v16h*)v1p;

        O0 = __builtin_amdgcn_wmma_f32_16x16x32_f16(false, pfrag, false, bv0, (short)0, O0, false, false);
        O1 = __builtin_amdgcn_wmma_f32_16x16x32_f16(false, pfrag, false, bv1, (short)0, O1, false, false);
    }

    sL[wv][myrow] = l;
    asm volatile("s_wait_dscnt 0x0" ::: "memory");

    // ---- write out: NCHW, channel = hd*24 + chead, position = q
    const int b = bh >> 1;
    #pragma unroll
    for (int r = 0; r < 8; ++r) {
        int M = r + hi * 8;
        float linv = 1.0f / sL[wv][M];
        int q = blockIdx.x * 128 + wv * 16 + M;
        int ch0 = ln16;                 // 0..15, all < 24 valid
        attn_out[((size_t)b * DIM + hd * CHEAD + ch0) * NPIX + q] = O0[r] * linv;
        if (ln16 < 8) {
            int ch1 = 16 + ln16;        // 16..23 valid
            attn_out[((size_t)b * DIM + hd * CHEAD + ch1) * NPIX + q] = O1[r] * linv;
        }
    }
}

// =====================================================================
// GELU(exact) gate: g[b][c][p] = gelu(f2[b][c][p]) * f2[b][96+c][p]
// =====================================================================
__global__ void gate_kernel(const float* __restrict__ f2,
                            float* __restrict__ g) {
    int t = blockIdx.x * blockDim.x + threadIdx.x;
    if (t >= BATCH * HID * NPIX) return;
    int p = t & (NPIX - 1);
    int c = (t >> 12) % HID;
    int b = t / (HID << 12);
    float y1 = f2[((size_t)b * HID2 + c) * NPIX + p];
    float y2 = f2[((size_t)b * HID2 + HID + c) * NPIX + p];
    float ge = y1 * 0.5f * (1.0f + erff(y1 * 0.70710678118654752f));
    g[t] = ge * y2;
}

// =====================================================================
// host launch
// =====================================================================
extern "C" void kernel_launch(void* const* d_in, const int* in_sizes, int n_in,
                              void* d_out, int out_size, void* d_ws, size_t ws_size,
                              hipStream_t stream) {
    (void)in_sizes; (void)n_in; (void)out_size; (void)ws_size;

    const float* x       = (const float*)d_in[0];
    const float* n1w     = (const float*)d_in[1];
    const float* n1b     = (const float*)d_in[2];
    const float* qkv_w   = (const float*)d_in[3];   // [144*48]
    const float* qkv_b   = (const float*)d_in[4];
    const float* qkvdw_w = (const float*)d_in[5];   // [144*9]
    const float* qkvdw_b = (const float*)d_in[6];
    const float* temp    = (const float*)d_in[7];   // [2]
    const float* proj_w  = (const float*)d_in[8];   // [48*48]
    const float* proj_b  = (const float*)d_in[9];
    const float* n2w     = (const float*)d_in[10];
    const float* n2b     = (const float*)d_in[11];
    const float* fi_w    = (const float*)d_in[12];  // [192*48]
    const float* fi_b    = (const float*)d_in[13];
    const float* fdw_w   = (const float*)d_in[14];  // [192*9]
    const float* fdw_b   = (const float*)d_in[15];
    const float* fo_w    = (const float*)d_in[16];  // [48*96]
    const float* fo_b    = (const float*)d_in[17];
    float* out = (float*)d_out;

    // workspace carve (256B aligned)
    char* ws = (char*)d_ws;
    size_t off = 0;
    auto carve = [&](size_t bytes) -> char* {
        char* p = ws + off;
        off += (bytes + 255) & ~(size_t)255;
        return p;
    };
    float*     xn    = (float*)    carve((size_t)BATCH * DIM  * NPIX * 4);
    float*     qkv   = (float*)    carve((size_t)BATCH * C3   * NPIX * 4);
    float*     qkvd  = (float*)    carve((size_t)BATCH * C3   * NPIX * 4);
    float*     aout  = (float*)    carve((size_t)BATCH * DIM  * NPIX * 4);
    float*     x1    = (float*)    carve((size_t)BATCH * DIM  * NPIX * 4);
    float*     xn2   = (float*)    carve((size_t)BATCH * DIM  * NPIX * 4);
    float*     f1    = (float*)    carve((size_t)BATCH * HID2 * NPIX * 4);
    float*     f2    = (float*)    carve((size_t)BATCH * HID2 * NPIX * 4);
    float*     g     = (float*)    carve((size_t)BATCH * HID  * NPIX * 4);
    _Float16*  Qp    = (_Float16*) carve((size_t)BH * NPIX * CPAD * 2);
    _Float16*  Kp    = (_Float16*) carve((size_t)BH * NPIX * CPAD * 2);
    _Float16*  Vt    = (_Float16*) carve((size_t)BH * CPAD * NPIX * 2);

    const int TPB = 256;

    // 1) LN1
    ln_kernel<<<(BATCH * NPIX + TPB - 1) / TPB, TPB, 0, stream>>>(x, n1w, n1b, xn);
    // 2) qkv 1x1 conv (48 -> 144)
    {
        int total = BATCH * C3 * NPIX;
        conv1x1_kernel<<<(total + TPB - 1) / TPB, TPB, 0, stream>>>(
            xn, qkv_w, qkv_b, nullptr, qkv, DIM, C3, total);
    }
    // 3) depthwise 3x3 on 144 channels
    {
        int total = BATCH * C3 * NPIX;
        dw3_kernel<<<(total + TPB - 1) / TPB, TPB, 0, stream>>>(
            qkv, qkvdw_w, qkvdw_b, qkvd, C3, total);
    }
    // 4) l2norm + pack f16
    pack_kernel<<<(BH * NPIX + TPB - 1) / TPB, TPB, 0, stream>>>(qkvd, Qp, Kp, Vt);
    // 5) flash attention (WMMA)
    attn_kernel<<<dim3(NPIX / 128, BH), 256, 0, stream>>>(Qp, Kp, Vt, temp, aout);
    // 6) proj 1x1 (48 -> 48) + residual with original x
    {
        int total = BATCH * DIM * NPIX;
        conv1x1_kernel<<<(total + TPB - 1) / TPB, TPB, 0, stream>>>(
            aout, proj_w, proj_b, x, x1, DIM, DIM, total);
    }
    // 7) LN2
    ln_kernel<<<(BATCH * NPIX + TPB - 1) / TPB, TPB, 0, stream>>>(x1, n2w, n2b, xn2);
    // 8) fi 1x1 (48 -> 192)
    {
        int total = BATCH * HID2 * NPIX;
        conv1x1_kernel<<<(total + TPB - 1) / TPB, TPB, 0, stream>>>(
            xn2, fi_w, fi_b, nullptr, f1, DIM, HID2, total);
    }
    // 9) depthwise 3x3 on 192 channels
    {
        int total = BATCH * HID2 * NPIX;
        dw3_kernel<<<(total + TPB - 1) / TPB, TPB, 0, stream>>>(
            f1, fdw_w, fdw_b, f2, HID2, total);
    }
    // 10) GELU gate
    {
        int total = BATCH * HID * NPIX;
        gate_kernel<<<(total + TPB - 1) / TPB, TPB, 0, stream>>>(f2, g);
    }
    // 11) fo 1x1 (96 -> 48) + residual with x1 -> d_out
    {
        int total = BATCH * DIM * NPIX;
        conv1x1_kernel<<<(total + TPB - 1) / TPB, TPB, 0, stream>>>(
            g, fo_w, fo_b, x1, out, HID, DIM, total);
    }
}
